// MyModel_87522843560581
// MI455X (gfx1250) — matrix operand order repaired and checked by reading, not compiled, not run on hardware
//
#include <hip/hip_runtime.h>
#include <hip/hip_bf16.h>

#define BB    256
#define TT    200
#define DD    311
#define UU    256
#define NOUT  19
#define MROWS (BB * TT)   // 51200 (b*T + t row index, matches reference layout)
#define DPAD  320         // 311 padded to multiple of 32 for K-loop
#define N3U   768
#define GRU_BLOCKS_PER_DIR 32

typedef _Float16 v16h __attribute__((ext_vector_type(16)));
typedef _Float16 v8h  __attribute__((ext_vector_type(8)));
typedef float    v8f  __attribute__((ext_vector_type(8)));

static __device__ __forceinline__ v16h cat8(v8h lo, v8h hi) {
  return __builtin_shufflevector(lo, hi, 0,1,2,3,4,5,6,7,8,9,10,11,12,13,14,15);
}

// A-frag 16x32 f16 from row-major [rows x ldk]; lane&15 = row-in-tile, lane>>4
// selects K-halves {0..7,16..23} vs {8..15,24..31}.  Two 16B contiguous loads.
static __device__ __forceinline__ v16h wmma_load_a(const _Float16* base, int ldk,
                                                   int row, int kc, int lane) {
  const _Float16* q = base + (size_t)row * ldk + kc + (lane >> 4) * 8;
  return cat8(*(const v8h*)q, *(const v8h*)(q + 16));
}

// B-frag 32x16 f16 from TRANSPOSED weights [N x ldk]; lane&15 = col,
// lane>>4 selects K-halves {0..7,16..23} vs {8..15,24..31}.  Two 16B loads.
static __device__ __forceinline__ v16h wmma_load_bT(const _Float16* baseT, int ldk,
                                                    int col, int kc, int lane) {
  const _Float16* q = baseT + (size_t)col * ldk + kc + (lane >> 4) * 8;
  return cat8(*(const v8h*)q, *(const v8h*)(q + 16));
}

static __device__ __forceinline__ v8f wmma32(v16h a, v16h b, v8f c) {
  return __builtin_amdgcn_wmma_f32_16x16x32_f16(false, a, false, b, (short)0, c,
                                                false, false);
}

// ---------------------------------------------------------------------------
// Zero the cross-block sync counters (graph-replay safe re-init each call)
__global__ void k_sync_init(int* __restrict__ sync) {
  if (threadIdx.x < 64) sync[threadIdx.x] = 0;
}

// ---------------------------------------------------------------------------
// x [MROWS x 311] f32 -> xh [MROWS x 320] f16 (zero-padded K) + mask=any(x!=0)
__global__ void k_prep_x(const float* __restrict__ x, _Float16* __restrict__ xh,
                         int* __restrict__ mask) {
  const int r = blockIdx.x;
  const float* xr = x + (size_t)r * DD;
  _Float16* xo = xh + (size_t)r * DPAD;
  int nz = 0;
  for (int d = threadIdx.x; d < DPAD; d += blockDim.x) {
    float v = (d < DD) ? xr[d] : 0.f;
    nz |= (v != 0.f);
    xo[d] = (_Float16)v;
  }
  __shared__ int s[8];
  const int wid = threadIdx.x >> 5;
  int anyv = __any(nz);
  if ((threadIdx.x & 31) == 0) s[wid] = anyv;
  __syncthreads();
  if (threadIdx.x == 0) {
    int m = 0;
    for (int i = 0; i < 8; ++i) m |= s[i];
    mask[r] = m;
  }
}

// ---------------------------------------------------------------------------
// Convert + TRANSPOSE weights to f16:  k[311x768] -> khT[768x320] (K padded),
// rk[256x768] -> rkT[768x256].  blockIdx.y selects which matrix.
__global__ void k_prep_w(const float* __restrict__ kf, const float* __restrict__ kb,
                         const float* __restrict__ rf, const float* __restrict__ rb,
                         _Float16* __restrict__ khTf, _Float16* __restrict__ khTb,
                         _Float16* __restrict__ rkTf, _Float16* __restrict__ rkTb) {
  const int n = blockIdx.x;        // output column 0..767
  const int which = blockIdx.y;    // 0..3
  const float* src; _Float16* dst; int kdim, ksrc;
  if (which == 0)      { src = kf; dst = khTf; kdim = DPAD; ksrc = DD; }
  else if (which == 1) { src = kb; dst = khTb; kdim = DPAD; ksrc = DD; }
  else if (which == 2) { src = rf; dst = rkTf; kdim = UU;   ksrc = UU; }
  else                 { src = rb; dst = rkTb; kdim = UU;   ksrc = UU; }
  for (int k = threadIdx.x; k < kdim; k += blockDim.x) {
    float v = (k < ksrc) ? src[(size_t)k * N3U + n] : 0.f;
    dst[(size_t)n * kdim + k] = (_Float16)v;
  }
}

// ---------------------------------------------------------------------------
// Input projection GEMM: xp_dir = xh @ k_dir + b_dir[0], stored f16.
// One wave computes a 32x32 tile (2x2 WMMA frags, A/B frag reuse).
__global__ void k_gemm_xp(const _Float16* __restrict__ xh,
                          const _Float16* __restrict__ khTf,
                          const _Float16* __restrict__ khTb,
                          const float* __restrict__ b_f,
                          const float* __restrict__ b_b,
                          _Float16* __restrict__ xpf,
                          _Float16* __restrict__ xpb) {
  const int dir = blockIdx.y;
  const _Float16* khT = dir ? khTb : khTf;
  const float* bias = dir ? b_b : b_f;          // row 0 of [2 x 768]
  _Float16* xp = dir ? xpb : xpf;

  const int lane = threadIdx.x & 31;
  const int wid  = threadIdx.x >> 5;
  const int tileIdx = blockIdx.x * 8 + wid;     // 0..38399
  const int tm = (tileIdx / 24) * 32;           // 1600 row-tiles of 32
  const int tn = (tileIdx % 24) * 32;           // 24 col-tiles of 32
  const int half = lane >> 4;
  const int l15  = lane & 15;

  v8f c00 = {0,0,0,0,0,0,0,0}, c01 = c00, c10 = c00, c11 = c00;
  for (int kc = 0; kc < DPAD; kc += 32) {
    v16h a0 = wmma_load_a(xh, DPAD, tm + l15,      kc, lane);
    v16h a1 = wmma_load_a(xh, DPAD, tm + 16 + l15, kc, lane);
    v16h b0 = wmma_load_bT(khT, DPAD, tn + l15,      kc, lane);
    v16h b1 = wmma_load_bT(khT, DPAD, tn + 16 + l15, kc, lane);
    c00 = wmma32(a0, b0, c00);
    c01 = wmma32(a0, b1, c01);
    c10 = wmma32(a1, b0, c10);
    c11 = wmma32(a1, b1, c11);
  }
  const int n0 = tn + l15, n1 = tn + 16 + l15;
  const float bv0 = bias[n0], bv1 = bias[n1];
#pragma unroll
  for (int i = 0; i < 8; ++i) {
    const int m0 = tm + half * 8 + i;   // D-frag: m = 8*half + i, n = lane&15
    const int m1 = m0 + 16;
    xp[(size_t)m0 * N3U + n0] = (_Float16)(c00[i] + bv0);
    xp[(size_t)m0 * N3U + n1] = (_Float16)(c01[i] + bv1);
    xp[(size_t)m1 * N3U + n0] = (_Float16)(c10[i] + bv0);
    xp[(size_t)m1 * N3U + n1] = (_Float16)(c11[i] + bv1);
  }
}

// ---------------------------------------------------------------------------
// Persistent GRU recurrence.  64 blocks: blocks [0,32) = forward, [32,64) =
// backward (directions run concurrently).  Each wave owns one 16x16 (b,u)
// block and computes ALL THREE gate columns for it (z at n=u, r at n=256+u,
// h~ at n=512+u), so one device-scope barrier per timestep suffices.
// Per-step hot data (rk f16 = 393KB/dir, h = 131KB/dir) is L2-resident.
__global__ void k_gru(const _Float16* __restrict__ xpf,
                      const _Float16* __restrict__ xpb,
                      const _Float16* __restrict__ rkTf,
                      const _Float16* __restrict__ rkTb,
                      const float* __restrict__ b_f,
                      const float* __restrict__ b_b,
                      const int* __restrict__ mask,
                      _Float16* __restrict__ hsf,
                      _Float16* __restrict__ hsb,
                      int* __restrict__ sync) {
  const int dir = blockIdx.x / GRU_BLOCKS_PER_DIR;
  const int blk = blockIdx.x % GRU_BLOCKS_PER_DIR;
  const _Float16* xp  = dir ? xpb : xpf;
  const _Float16* rkT = dir ? rkTb : rkTf;
  _Float16* hs        = dir ? hsb : hsf;
  const float* b1 = (dir ? b_b : b_f) + N3U;    // recurrent bias row
  int* ctr = sync + dir;

  const int lane = threadIdx.x & 31;
  const int wid  = threadIdx.x >> 5;
  const int w    = blk * 8 + wid;               // 0..255 waves per direction
  const int b0   = (w >> 4) * 16;               // batch tile
  const int u0   = (w & 15) * 16;               // unit tile
  const int half = lane >> 4;
  const int l15  = lane & 15;
  const int u_lane = u0 + l15;

  const float bz = b1[u_lane];
  const float br = b1[UU + u_lane];
  const float bh = b1[2 * UU + u_lane];

  // fixed B-frag row pointers (transposed rk), one per gate
  const _Float16* bpz = rkT + (size_t)(0 * UU + u_lane) * UU + half * 8;
  const _Float16* bpr = rkT + (size_t)(1 * UU + u_lane) * UU + half * 8;
  const _Float16* bph = rkT + (size_t)(2 * UU + u_lane) * UU + half * 8;

  for (int s = 0; s < TT; ++s) {
    const int t  = dir ? (TT - 1 - s) : s;
    const int tp = dir ? (t + 1) : (t - 1);
    v8f cz = {0,0,0,0,0,0,0,0}, cr = cz, ch = cz;
    if (s > 0) {                                 // uniform branch: EXEC all-1s
      const _Float16* ap = hs + ((size_t)(b0 + l15) * TT + tp) * UU + half * 8;
#pragma unroll
      for (int kc = 0; kc < UU; kc += 32) {
        v16h a  = cat8(*(const v8h*)(ap + kc), *(const v8h*)(ap + kc + 16));
        v16h vz = cat8(*(const v8h*)(bpz + kc), *(const v8h*)(bpz + kc + 16));
        cz = wmma32(a, vz, cz);
        v16h vr = cat8(*(const v8h*)(bpr + kc), *(const v8h*)(bpr + kc + 16));
        cr = wmma32(a, vr, cr);
        v16h vh = cat8(*(const v8h*)(bph + kc), *(const v8h*)(bph + kc + 16));
        ch = wmma32(a, vh, ch);
      }
    }
#pragma unroll
    for (int i = 0; i < 8; ++i) {
      const int bi = b0 + half * 8 + i;          // D-frag row = batch index
      const size_t rbt = (size_t)bi * TT + t;
      const float xz = (float)xp[rbt * N3U + u_lane];
      const float xr = (float)xp[rbt * N3U + UU + u_lane];
      const float xc = (float)xp[rbt * N3U + 2 * UU + u_lane];
      const float rz = cz[i] + bz;
      const float rr = cr[i] + br;
      const float rh = ch[i] + bh;
      const float z  = 1.f / (1.f + __expf(-(xz + rz)));
      const float r  = 1.f / (1.f + __expf(-(xr + rr)));
      const float hc = tanhf(xc + r * rh);       // reset_after: r * (h@rk + b1)
      float hp = 0.f;
      if (s > 0) hp = (float)hs[((size_t)bi * TT + tp) * UU + u_lane];
      float hn = z * hp + (1.f - z) * hc;
      hn = mask[rbt] ? hn : hp;                  // masked step carries state
      hs[rbt * UU + u_lane] = (_Float16)hn;
    }
    // per-direction device-scope barrier between timesteps
    __syncthreads();
    if (threadIdx.x == 0) {
      __threadfence();
      __hip_atomic_fetch_add(ctr, 1, __ATOMIC_ACQ_REL, __HIP_MEMORY_SCOPE_AGENT);
      const int target = GRU_BLOCKS_PER_DIR * (s + 1);
      while (__hip_atomic_load(ctr, __ATOMIC_ACQUIRE, __HIP_MEMORY_SCOPE_AGENT) < target)
        __builtin_amdgcn_s_sleep(1);
    }
    __syncthreads();
  }
}

// ---------------------------------------------------------------------------
// TimeDistributed Dense(19) + softmax.  One wave per (b,t) row; lane = output
// class; h broadcast via shuffles; shuffle-reduced softmax.  Memory-bound.
__global__ void k_dense(const _Float16* __restrict__ hsf,
                        const _Float16* __restrict__ hsb,
                        const float* __restrict__ wd,
                        const float* __restrict__ bd,
                        float* __restrict__ out) {
  const int lane = threadIdx.x & 31;
  const int wid  = threadIdx.x >> 5;
  const int r = blockIdx.x * 8 + wid;           // (b*T + t) row
  const _Float16* hf = hsf + (size_t)r * UU;
  const _Float16* hb = hsb + (size_t)r * UU;
  float acc = 0.f;
  for (int u0 = 0; u0 < UU; u0 += 32) {
    float hv = (float)hf[u0 + lane];
#pragma unroll
    for (int j = 0; j < 32; ++j) {
      float h = __shfl(hv, j, 32);
      float wv = (lane < NOUT) ? wd[(size_t)(u0 + j) * NOUT + lane] : 0.f;
      acc += h * wv;
    }
  }
  for (int u0 = 0; u0 < UU; u0 += 32) {
    float hv = (float)hb[u0 + lane];
#pragma unroll
    for (int j = 0; j < 32; ++j) {
      float h = __shfl(hv, j, 32);
      float wv = (lane < NOUT) ? wd[(size_t)(UU + u0 + j) * NOUT + lane] : 0.f;
      acc += h * wv;
    }
  }
  float logit = (lane < NOUT) ? (acc + bd[lane]) : -3.4e38f;
  float mx = logit;
#pragma unroll
  for (int o = 16; o > 0; o >>= 1) mx = fmaxf(mx, __shfl_xor(mx, o, 32));
  float e = (lane < NOUT) ? __expf(logit - mx) : 0.f;
  float sum = e;
#pragma unroll
  for (int o = 16; o > 0; o >>= 1) sum += __shfl_xor(sum, o, 32);
  if (lane < NOUT) out[(size_t)r * NOUT + lane] = e / sum;
}

// ---------------------------------------------------------------------------
extern "C" void kernel_launch(void* const* d_in, const int* in_sizes, int n_in,
                              void* d_out, int out_size, void* d_ws, size_t ws_size,
                              hipStream_t stream) {
  (void)in_sizes; (void)n_in; (void)out_size; (void)ws_size;
  const float* x    = (const float*)d_in[0];
  const float* k_f  = (const float*)d_in[1];
  const float* rk_f = (const float*)d_in[2];
  const float* b_f  = (const float*)d_in[3];
  const float* k_b  = (const float*)d_in[4];
  const float* rk_b = (const float*)d_in[5];
  const float* b_b  = (const float*)d_in[6];
  const float* w_d  = (const float*)d_in[7];
  const float* b_d  = (const float*)d_in[8];
  float* out = (float*)d_out;

  char* p = (char*)d_ws;
  size_t off = 0;
  auto take = [&](size_t nbytes) -> void* {
    void* q = p + off;
    off = (off + nbytes + 255) & ~(size_t)255;
    return q;
  };
  int*      syncp = (int*)take(256);
  _Float16* xh    = (_Float16*)take((size_t)MROWS * DPAD * sizeof(_Float16));
  _Float16* khTf  = (_Float16*)take((size_t)N3U * DPAD * sizeof(_Float16));
  _Float16* khTb  = (_Float16*)take((size_t)N3U * DPAD * sizeof(_Float16));
  _Float16* rkTf  = (_Float16*)take((size_t)N3U * UU * sizeof(_Float16));
  _Float16* rkTb  = (_Float16*)take((size_t)N3U * UU * sizeof(_Float16));
  _Float16* xpf   = (_Float16*)take((size_t)MROWS * N3U * sizeof(_Float16));
  _Float16* xpb   = (_Float16*)take((size_t)MROWS * N3U * sizeof(_Float16));
  _Float16* hsf   = (_Float16*)take((size_t)MROWS * UU * sizeof(_Float16));
  _Float16* hsb   = (_Float16*)take((size_t)MROWS * UU * sizeof(_Float16));
  int*      maskp = (int*)take((size_t)MROWS * sizeof(int));

  k_sync_init<<<1, 64, 0, stream>>>(syncp);
  k_prep_x<<<MROWS, 256, 0, stream>>>(x, xh, maskp);
  k_prep_w<<<dim3(N3U, 4), 64, 0, stream>>>(k_f, k_b, rk_f, rk_b,
                                            khTf, khTb, rkTf, rkTb);
  k_gemm_xp<<<dim3(4800, 2), 256, 0, stream>>>(xh, khTf, khTb, b_f, b_b,
                                               xpf, xpb);
  k_gru<<<2 * GRU_BLOCKS_PER_DIR, 256, 0, stream>>>(xpf, xpb, rkTf, rkTb,
                                                    b_f, b_b, maskp,
                                                    hsf, hsb, syncp);
  k_dense<<<MROWS / 8, 256, 0, stream>>>(hsf, hsb, w_d, b_d, out);
}
